// LSTMAgent_52819507806639
// MI455X (gfx1250) — compile-verified, matrix operand order checked by reading, not run on hardware
//
#include <hip/hip_runtime.h>
#include <hip/hip_bf16.h>
#include <math.h>

typedef __attribute__((ext_vector_type(16))) _Float16 v16h;
typedef __attribute__((ext_vector_type(8)))  _Float16 h8;
typedef __attribute__((ext_vector_type(8)))  float    v8f;

#define T_STEPS 64
#define BATCH   32
#define NSEQ    (T_STEPS * BATCH)   // 2048
#define HID     128
#define GATES   512                  // 4*HID
#define INP     544                  // 532 padded to multiple of 32

// ---------------- workspace layout (bytes, all 256B aligned) ----------------
enum : size_t {
  OFF_IMG  = 0,            // 2048*84*84*3 f16      = 86,704,128
  OFF_W1T  = 86704128,     // 32*192 f16
  OFF_W2T  = 86716416,     // 64*512 f16
  OFF_W3T  = 86781952,     // 64*576 f16
  OFF_WFCT = 86855680,     // 512*3136 f16
  OFF_WIHT = 90066944,     // 512*544 f16
  OFF_WHH  = 90624000,     // 512*128 f16
  OFF_A1   = 90755072,     // 2048*400*32 f16
  OFF_A2   = 143183872,    // 2048*81*64 f16
  OFF_A3   = 164417536,    // 2048*3136 f16
  OFF_Z    = 177262592,    // 2048*544 f16
  OFF_GX   = 179490816,    // 2048*512 f32
  OFF_HS   = 183685120     // 2048*128 f32
};

// ---------------- WMMA helpers ----------------
__device__ __forceinline__ v8f wmma16(v16h a, v16h b, v8f c) {
  // D = A(16x32 f16) * B(32x16 f16) + C(16x16 f32)
  return __builtin_amdgcn_wmma_f32_16x16x32_f16(false, a, false, b, (short)0, c,
                                                false, false);
}

// A/B fragment pointer for the 16-bit 16x32 layout:
// lanes 0-15: row M=lane,  K = {kb..kb+7, 16+kb..16+kb+7} with kb=0
// lanes 16-31: row M=lane-16, kb=8
__device__ __forceinline__ const _Float16* frag_ptr(const _Float16* base, int stride,
                                                    int k0, int lane) {
  int row = lane & 15;
  int kb  = (lane >> 4) << 3;
  return base + row * stride + k0 + kb;
}

__device__ __forceinline__ v16h load_frag(const _Float16* q) {
  h8 lo = *(const h8*)q;         // K = kb .. kb+7
  h8 hi = *(const h8*)(q + 16);  // K = 16+kb .. 16+kb+7
  return __builtin_shufflevector(lo, hi, 0,1,2,3,4,5,6,7,8,9,10,11,12,13,14,15);
}

// ---------------- precision / layout conversion kernels ----------------
__global__ void k_cvt_f16(const float* __restrict__ in, _Float16* __restrict__ out, int n) {
  int i = blockIdx.x * blockDim.x + threadIdx.x;
  if (i < n) out[i] = (_Float16)in[i];
}

// in: [K][N] f32  ->  out: [N][K] f16   (for conv/FC weights as WMMA B operand)
__global__ void k_cvt_transpose(const float* __restrict__ in, _Float16* __restrict__ out,
                                int K, int N) {
  int i = blockIdx.x * blockDim.x + threadIdx.x;
  if (i < K * N) {
    int k = i / N, n = i - k * N;
    out[(size_t)n * K + k] = (_Float16)in[i];
  }
}

// Wih: [512][532] f32 -> [512][544] f16 zero padded
__global__ void k_cvt_wih(const float* __restrict__ in, _Float16* __restrict__ out) {
  int i = blockIdx.x * blockDim.x + threadIdx.x;
  if (i < GATES * INP) {
    int n = i / INP, k = i - n * INP;
    out[i] = (k < 532) ? (_Float16)in[n * 532 + k] : (_Float16)0.f;
  }
}

// ---------------- implicit-GEMM conv ----------------
// One wave per (image, 16-row m-tile). All COUT tiles (NR accumulators) per wave:
// A (im2col) staged once into LDS for the full K; B fragments read directly from
// global (weights are [COUT][K], K % 32 == 0, L2-resident). MR=1 keeps register
// pressure low enough to avoid spills with the fully unrolled K loop.
template <int OH, int OW, int KH, int KW, int CIN, int COUT, int STRIDE, int IH, int IW,
          int VEC>
__global__ void k_conv_wmma(const _Float16* __restrict__ in,   // [N][IH][IW][CIN]
                            const _Float16* __restrict__ wT,   // [COUT][K]
                            const float* __restrict__ bias,
                            _Float16* __restrict__ out) {      // [N][OH*OW][COUT]
  constexpr int K  = KH * KW * CIN;
  constexpr int M  = OH * OW;
  constexpr int MT = (M + 15) / 16;
  constexpr int NR = COUT / 16;
  constexpr int WPB = 2;
  __shared__ __align__(16) _Float16 sA[WPB][16 * K];

  int wslot = threadIdx.x >> 5;
  int lane  = threadIdx.x & 31;
  int wid   = blockIdx.x * WPB + wslot;
  if (wid >= NSEQ * MT) return;
  int img = wid / MT, mt = wid - img * MT;

  const _Float16* src = in + (size_t)img * (IH * IW * CIN);
  _Float16* ldsA = sA[wslot];

  // ---- stage A: 16 output rows x K (im2col), zero rows past M ----
  if (VEC) {
    // CIN % 8 == 0: 8-half runs contiguous and 16B aligned in NHWC memory
    for (int idx = lane; idx < 16 * (K / 8); idx += 32) {
      int r  = idx / (K / 8);
      int k8 = (idx - r * (K / 8)) * 8;
      int m  = mt * 16 + r;
      h8 v = {};
      if (m < M) {
        int oy = m / OW, ox = m - oy * OW;
        int kh = k8 / (KW * CIN); int r2 = k8 - kh * (KW * CIN);
        int kw = r2 / CIN;        int ci = r2 - kw * CIN;
        v = *(const h8*)(src + ((oy * STRIDE + kh) * IW + (ox * STRIDE + kw)) * CIN + ci);
      }
      *(h8*)(ldsA + r * K + k8) = v;
    }
  } else {
    for (int idx = lane; idx < 16 * K; idx += 32) {
      int r = idx / K, kk = idx - r * K;
      int m = mt * 16 + r;
      _Float16 v = (_Float16)0.f;
      if (m < M) {
        int oy = m / OW, ox = m - oy * OW;
        int kh = kk / (KW * CIN); int r2 = kk - kh * (KW * CIN);
        int kw = r2 / CIN;        int ci = r2 - kw * CIN;
        v = src[((oy * STRIDE + kh) * IW + (ox * STRIDE + kw)) * CIN + ci];
      }
      ldsA[idx] = v;
    }
  }
  // per-wave LDS ops are in-order; no cross-wave barrier needed

  v8f acc[NR];
#pragma unroll
  for (int j = 0; j < NR; ++j) acc[j] = {};

  for (int k0 = 0; k0 < K; k0 += 32) {
    v16h af = load_frag(frag_ptr(ldsA, K, k0, lane));
#pragma unroll
    for (int j = 0; j < NR; ++j) {
      v16h bf = load_frag(frag_ptr(wT + (size_t)(j * 16) * K, K, k0, lane));
      acc[j] = wmma16(af, bf, acc[j]);
    }
  }

  // epilogue: bias + ReLU + f16 store. D layout: n=lane&15, m = r + 8*(lane>>4)
  int n = lane & 15, mo = (lane >> 4) * 8;
#pragma unroll
  for (int j = 0; j < NR; ++j) {
    float bc = bias[j * 16 + n];
#pragma unroll
    for (int r = 0; r < 8; ++r) {
      int m = mt * 16 + mo + r;
      if (m < M) {
        float v = acc[j][r] + bc;
        v = v > 0.f ? v : 0.f;
        out[(size_t)img * (M * COUT) + (size_t)m * COUT + j * 16 + n] = (_Float16)v;
      }
    }
  }
}

// ---------------- generic WMMA GEMM: C = relu(A[M][K] * B[N][K]^T + bias) ----------------
// No LDS: A and B fragments loaded directly from global (L2-resident), K % 32 == 0,
// M multiple of 32, N multiple of 64. Each wave computes a 32x64 strip: per k-step
// 2 A frags + 4 B frags + 8 independent WMMAs (covers WMMA->WMMA RAW latency).
// k-loop kept rolled (unroll 1) so live fragments stay bounded -> no spills.
template <bool RELU, bool OUT16>
__global__ void k_gemm_wmma(const _Float16* __restrict__ A, const _Float16* __restrict__ Bm,
                            const float* __restrict__ bias, const float* __restrict__ bias2,
                            _Float16* __restrict__ o16, float* __restrict__ o32,
                            int M, int N, int K, int ldo) {
  constexpr int MR = 2, NR = 4;
  int lane = threadIdx.x & 31;
  int wid  = blockIdx.x * (blockDim.x >> 5) + (threadIdx.x >> 5);
  int NG = N / (16 * NR);
  int mg = wid / NG, ng = wid - mg * NG;
  if (mg >= M / (16 * MR)) return;
  int m0 = mg * 16 * MR;
  int n0 = ng * 16 * NR;
  const _Float16* Abase = A + (size_t)m0 * K;

  v8f acc[MR][NR];
#pragma unroll
  for (int i = 0; i < MR; ++i)
#pragma unroll
    for (int j = 0; j < NR; ++j) acc[i][j] = {};

#pragma unroll 1
  for (int k0 = 0; k0 < K; k0 += 32) {
    if (k0 + 512 < K) {
      // keep L0 warm for upcoming k-steps (global_prefetch_b8)
      __builtin_prefetch((const void*)(Abase + (size_t)lane * K + k0 + 512), 0, 1);
      __builtin_prefetch((const void*)(Bm + (size_t)(n0 + lane) * K + k0 + 512), 0, 1);
      __builtin_prefetch((const void*)(Bm + (size_t)(n0 + 32 + lane) * K + k0 + 512), 0, 1);
    }
    v16h af[MR];
#pragma unroll
    for (int i = 0; i < MR; ++i)
      af[i] = load_frag(frag_ptr(Abase + (size_t)(i * 16) * K, K, k0, lane));
#pragma unroll
    for (int j = 0; j < NR; ++j) {
      v16h bf = load_frag(frag_ptr(Bm + (size_t)(n0 + j * 16) * K, K, k0, lane));
#pragma unroll
      for (int i = 0; i < MR; ++i) acc[i][j] = wmma16(af[i], bf, acc[i][j]);
    }
  }

  int n = lane & 15, mo = (lane >> 4) * 8;
#pragma unroll
  for (int i = 0; i < MR; ++i) {
#pragma unroll
    for (int j = 0; j < NR; ++j) {
      int col = n0 + j * 16 + n;
      float bc = (bias ? bias[col] : 0.f) + (bias2 ? bias2[col] : 0.f);
#pragma unroll
      for (int r = 0; r < 8; ++r) {
        int gm = m0 + i * 16 + mo + r;
        float v = acc[i][j][r] + bc;
        if (RELU) v = v > 0.f ? v : 0.f;
        if (OUT16) o16[(size_t)gm * ldo + col] = (_Float16)v;
        else       o32[(size_t)gm * ldo + col] = v;
      }
    }
  }
}

// ---------------- one-hot(position) into z[:,512:544] ----------------
__global__ void k_onehot(const int* __restrict__ pos, _Float16* __restrict__ z) {
  int gid = blockIdx.x * blockDim.x + threadIdx.x;
  if (gid >= NSEQ * 32) return;
  int row = gid >> 5, j = gid & 31;
  float v = 0.f;
  if (j < 10)      v = (pos[row * 2 + 0] == j)        ? 1.f : 0.f;
  else if (j < 20) v = (pos[row * 2 + 1] == (j - 10)) ? 1.f : 0.f;
  z[(size_t)row * INP + 512 + j] = (_Float16)v;
}

// ---------------- LSTM recurrence: one workgroup, 16 waves, WMMA per step ----------------
// Tiles processed sequentially (4 per wave) so only ~2 fragments are live at a time:
// all 16 waves share one WGP, so per-wave VGPR budget is limited — this shape is the
// one that codegens to clause'd global_load_b128 -> v_wmma with no scratch spills.
__global__ void k_lstm(const float* __restrict__ gx,       // [T*B][512] = x@Wih^T + bih + bhh
                       const _Float16* __restrict__ whhT,  // [512][128] f16
                       const float* __restrict__ h0, const float* __restrict__ c0,
                       const int* __restrict__ done,
                       float* __restrict__ Hs,             // [T*B][128]
                       float* __restrict__ houtF, float* __restrict__ coutF) {
  __shared__ __align__(16) _Float16 h16[BATCH * HID];    // 8 KB
  __shared__ __align__(16) float    cF[BATCH * HID];     // 16 KB
  __shared__ __align__(16) _Float16 g16[BATCH * GATES];  // 32 KB
  int tid = threadIdx.x, lane = tid & 31, w = tid >> 5;

  for (int i = tid; i < BATCH * HID; i += 512) {
    h16[i] = (_Float16)h0[i];
    cF[i]  = c0[i];
  }
  __syncthreads();

  for (int t = 0; t < T_STEPS; ++t) {
    // mask carry by done
    for (int i = tid; i < BATCH * HID; i += 512) {
      int b = i >> 7;
      float m = 1.0f - (float)done[t * BATCH + b];
      h16[i] = (_Float16)((float)h16[i] * m);
      cF[i] *= m;
    }
    __syncthreads();
    // gates = gx[t] + h @ Whh^T ; 64 tiles (2 M-tiles x 32 N-tiles), 4 per wave
    for (int tt = w; tt < 64; tt += 16) {
      int mt = tt & 1, nt = tt >> 1;
      int m0 = mt * 16, n0 = nt * 16;
      int n = lane & 15, mo = (lane >> 4) * 8;
      v8f acc = {};
#pragma unroll
      for (int r = 0; r < 8; ++r)
        acc[r] = gx[(size_t)(t * BATCH + m0 + mo + r) * GATES + n0 + n];
#pragma unroll
      for (int ks = 0; ks < HID / 32; ++ks) {
        v16h af = load_frag(frag_ptr(h16 + m0 * HID, HID, ks * 32, lane));
        v16h bf = load_frag(frag_ptr(whhT + n0 * HID, HID, ks * 32, lane));
        acc = wmma16(af, bf, acc);
      }
#pragma unroll
      for (int r = 0; r < 8; ++r)
        g16[(m0 + mo + r) * GATES + n0 + n] = (_Float16)acc[r];
    }
    __syncthreads();
    // element-wise cell update
    for (int i = tid; i < BATCH * HID; i += 512) {
      int b = i >> 7, hh = i & (HID - 1);
      const _Float16* gr = g16 + b * GATES;
      float gi = 1.f / (1.f + expf(-(float)gr[hh]));
      float gf = 1.f / (1.f + expf(-(float)gr[HID + hh]));
      float gg = tanhf((float)gr[2 * HID + hh]);
      float go = 1.f / (1.f + expf(-(float)gr[3 * HID + hh]));
      float cv = gf * cF[i] + gi * gg;
      float hv = go * tanhf(cv);
      cF[i]  = cv;
      h16[i] = (_Float16)hv;
      Hs[(size_t)(t * BATCH + b) * HID + hh] = hv;
    }
    __syncthreads();
  }
  for (int i = tid; i < BATCH * HID; i += 512) {
    houtF[i] = (float)h16[i];
    coutF[i] = cF[i];
  }
}

// ---------------- policy / value heads (tiny, VALU) ----------------
__global__ void k_heads(const float* __restrict__ Hs, const float* __restrict__ Wp,
                        const float* __restrict__ bp, const float* __restrict__ Wv,
                        const float* __restrict__ bv, float* __restrict__ out) {
  int gid = blockIdx.x * blockDim.x + threadIdx.x;
  int row = gid / 6, j = gid - row * 6;
  if (row >= NSEQ) return;
  const float* h = Hs + (size_t)row * HID;
  float s = 0.f;
  if (j < 5) {
    for (int k = 0; k < HID; ++k) s += h[k] * Wp[k * 5 + j];
    out[row * 5 + j] = s + bp[j];
  } else {
    for (int k = 0; k < HID; ++k) s += h[k] * Wv[k];
    out[NSEQ * 5 + row] = s + bv[0];
  }
}

// ---------------- launcher ----------------
extern "C" void kernel_launch(void* const* d_in, const int* in_sizes, int n_in,
                              void* d_out, int out_size, void* d_ws, size_t ws_size,
                              hipStream_t stream) {
  (void)in_sizes; (void)n_in; (void)out_size; (void)ws_size;
  const float* image = (const float*)d_in[0];
  const int*   pos   = (const int*)d_in[1];
  const int*   done  = (const int*)d_in[2];
  const float* h0    = (const float*)d_in[3];
  const float* c0    = (const float*)d_in[4];
  const float* W1    = (const float*)d_in[5];
  const float* b1    = (const float*)d_in[6];
  const float* W2    = (const float*)d_in[7];
  const float* b2    = (const float*)d_in[8];
  const float* W3    = (const float*)d_in[9];
  const float* b3    = (const float*)d_in[10];
  const float* Wfc   = (const float*)d_in[11];
  const float* bfc   = (const float*)d_in[12];
  const float* Wih   = (const float*)d_in[13];
  const float* Whh   = (const float*)d_in[14];
  const float* bih   = (const float*)d_in[15];
  const float* bhh   = (const float*)d_in[16];
  const float* Wp    = (const float*)d_in[17];
  const float* bp    = (const float*)d_in[18];
  const float* Wv    = (const float*)d_in[19];
  const float* bv    = (const float*)d_in[20];
  float* out = (float*)d_out;

  char* ws = (char*)d_ws;
  _Float16* img16 = (_Float16*)(ws + OFF_IMG);
  _Float16* w1T   = (_Float16*)(ws + OFF_W1T);
  _Float16* w2T   = (_Float16*)(ws + OFF_W2T);
  _Float16* w3T   = (_Float16*)(ws + OFF_W3T);
  _Float16* wfcT  = (_Float16*)(ws + OFF_WFCT);
  _Float16* wihT  = (_Float16*)(ws + OFF_WIHT);
  _Float16* whh16 = (_Float16*)(ws + OFF_WHH);
  _Float16* a1    = (_Float16*)(ws + OFF_A1);
  _Float16* a2    = (_Float16*)(ws + OFF_A2);
  _Float16* a3    = (_Float16*)(ws + OFF_A3);
  _Float16* z     = (_Float16*)(ws + OFF_Z);
  float*    gx    = (float*)(ws + OFF_GX);
  float*    Hs    = (float*)(ws + OFF_HS);

  // precision / layout conversion
  int nImg = NSEQ * 84 * 84 * 3;
  k_cvt_f16<<<(nImg + 255) / 256, 256, 0, stream>>>(image, img16, nImg);
  k_cvt_transpose<<<(192 * 32 + 255) / 256, 256, 0, stream>>>(W1, w1T, 192, 32);
  k_cvt_transpose<<<(512 * 64 + 255) / 256, 256, 0, stream>>>(W2, w2T, 512, 64);
  k_cvt_transpose<<<(576 * 64 + 255) / 256, 256, 0, stream>>>(W3, w3T, 576, 64);
  k_cvt_transpose<<<(3136 * 512 + 255) / 256, 256, 0, stream>>>(Wfc, wfcT, 3136, 512);
  k_cvt_wih<<<(GATES * INP + 255) / 256, 256, 0, stream>>>(Wih, wihT);
  k_cvt_f16<<<(GATES * HID + 255) / 256, 256, 0, stream>>>(Whh, whh16, GATES * HID);

  // CNN trunk (implicit GEMM, f16 WMMA, fused bias+ReLU); 2 waves / 64-thread block
  k_conv_wmma<20, 20, 8, 8, 3, 32, 4, 84, 84, 0><<<25600, 64, 0, stream>>>(img16, w1T, b1, a1);
  k_conv_wmma<9, 9, 4, 4, 32, 64, 2, 20, 20, 1><<<6144, 64, 0, stream>>>(a1, w2T, b2, a2);
  k_conv_wmma<7, 7, 3, 3, 64, 64, 1, 9, 9, 1><<<4096, 64, 0, stream>>>(a2, w3T, b3, a3);

  // FC (2048x3136x512) -> z[:, :512]; 512 waves, each a 32x64 strip
  k_gemm_wmma<true, true><<<128, 128, 0, stream>>>(a3, wfcT, bfc, nullptr, z, nullptr,
                                                   NSEQ, 512, 3136, INP);
  k_onehot<<<(NSEQ * 32 + 255) / 256, 256, 0, stream>>>(pos, z);

  // hoisted input gates: gx = z @ Wih^T + bih + bhh  (2048x544x512, f32 out)
  k_gemm_wmma<false, false><<<128, 128, 0, stream>>>(z, wihT, bih, bhh, nullptr, gx,
                                                     NSEQ, GATES, INP, GATES);

  // sequential recurrence (single WGP, LDS-resident state, WMMA per step)
  k_lstm<<<1, 512, 0, stream>>>(gx, whh16, h0, c0, done, Hs,
                                out + NSEQ * 5 + NSEQ,                // h at 12288
                                out + NSEQ * 5 + NSEQ + BATCH * HID); // c at 16384

  // heads: logits (2048x5) at out[0], v (2048) at out[10240]
  k_heads<<<(NSEQ * 6 + 127) / 128, 128, 0, stream>>>(Hs, Wp, bp, Wv, bv, out);
}
